// SIGReg_25649544692275
// MI455X (gfx1250) — matrix operand order repaired and checked by reading, not compiled, not run on hardware
//
#include <hip/hip_runtime.h>
#include <hip/hip_bf16.h>

typedef __attribute__((ext_vector_type(16))) _Float16 v16h;
typedef __attribute__((ext_vector_type(8)))  _Float16 v8h;
typedef __attribute__((ext_vector_type(4)))  _Float16 v4h;
typedef __attribute__((ext_vector_type(8)))  float    v8f;

#define N_ROWS 8192
#define D_DIM  768
#define P_DIM  1024
#define KNOTS  17
#define TILE_M 128  // block M tile (4 waves x 32 rows)
#define TILE_N 32   // block N tile (2 waves x 16 cols)
#define TILE_K 32
#define WAVE_M 32   // rows per wave (2 WMMAs)
#define LDSP   40   // padded LDS row stride in halves (80B: multiple of 16B)

// ---------------------------------------------------------------- zero scratch
__global__ void sig_zero_kernel(float* p, int n) {
    int i = blockIdx.x * blockDim.x + threadIdx.x;
    if (i < n) p[i] = 0.0f;
}

// ------------------------------------------------- column inverse norms of `a`
// a is (D_DIM, P_DIM) row-major; thread per column -> fully coalesced.
__global__ void sig_colnorm_kernel(const float* __restrict__ a,
                                   float* __restrict__ invn) {
    int p = blockIdx.x * blockDim.x + threadIdx.x;
    if (p >= P_DIM) return;
    float s = 0.0f;
    for (int i = 0; i < D_DIM; ++i) {
        float v = a[(size_t)i * P_DIM + p];
        s += v * v;
    }
    invn[p] = rsqrtf(s);
}

// --------------------------------------- fused WMMA GEMM + characteristic fn
// Block: 256 threads = 8 waves, arranged 4 (M) x 2 (N); each wave owns a
// 32x16 output tile -> 2 WMMAs per K chunk. Block tile: 128x32.
// Grid: (P_DIM/TILE_N, N_ROWS/TILE_M) = (32, 64).
// Normalization is applied post-GEMM: z @ (a*diag(invn)) == (z @ a)*diag(invn).
__global__ __launch_bounds__(256)
void sig_fused_kernel(const float* __restrict__ z,
                      const float* __restrict__ a,
                      const float* __restrict__ invn,
                      float* __restrict__ cosA,
                      float* __restrict__ sinA) {
    __shared__ _Float16 ldsZ[TILE_M * LDSP];   // z tile,  M x K, row-major
    __shared__ _Float16 ldsAT[TILE_N * LDSP];  // a tile,  N x K (transposed)
    __shared__ float lds_cos[TILE_N * KNOTS];
    __shared__ float lds_sin[TILE_N * KNOTS];

    const int tid  = threadIdx.x;
    const int lane = tid & 31;
    const int wave = tid >> 5;
    const int m_base = (wave & 3) * WAVE_M;  // wave's M offset in block tile
    const int n_base = (wave >> 2) * 16;     // wave's N offset in block tile
    const int p_block = blockIdx.x * TILE_N;
    const int m_block = blockIdx.y * TILE_M;

    for (int i = tid; i < TILE_N * KNOTS; i += 256) {
        lds_cos[i] = 0.0f;
        lds_sin[i] = 0.0f;
    }

    // staging coordinates: 128 rows x 8 float4-quads = 1024 slots, 4 per thread
    const int zr = tid >> 3;  // z row within tile (0..31; +32/64/96 per pass)
    const int zq = tid & 7;   // float4 quad within the 32-wide K chunk
    const int ak = tid >> 3;  // a K-row within chunk (0..31)
    const int aq = tid & 7;   // float4 quad across 32 columns

    v8f c0 = {};  // rows m_base .. m_base+15
    v8f c1 = {};  // rows m_base+16 .. m_base+31

    for (int kc = 0; kc < D_DIM; kc += TILE_K) {
        // ---- issue all global loads first (overlapped), then convert/store
        float4 zv[4];
        #pragma unroll
        for (int rr = 0; rr < 4; ++rr)
            zv[rr] = *reinterpret_cast<const float4*>(
                z + (size_t)(m_block + zr + rr * 32) * D_DIM + kc + zq * 4);
        const float4 av = *reinterpret_cast<const float4*>(
            a + (size_t)(kc + ak) * P_DIM + p_block + aq * 4);

        if (kc + TILE_K < D_DIM) {  // prefetch next K chunk toward the WGP
            #pragma unroll
            for (int rr = 0; rr < 4; ++rr)
                __builtin_prefetch(z + (size_t)(m_block + zr + rr * 32) * D_DIM +
                                   kc + TILE_K + zq * 4, 0, 3);
            __builtin_prefetch(a + (size_t)(kc + TILE_K + ak) * P_DIM +
                               p_block + aq * 4, 0, 3);
        }

        #pragma unroll
        for (int rr = 0; rr < 4; ++rr) {
            v4h hz = { (_Float16)zv[rr].x, (_Float16)zv[rr].y,
                       (_Float16)zv[rr].z, (_Float16)zv[rr].w };
            *reinterpret_cast<v4h*>(&ldsZ[(zr + rr * 32) * LDSP + zq * 4]) = hz;
        }
        ldsAT[(aq * 4 + 0) * LDSP + ak] = (_Float16)av.x;
        ldsAT[(aq * 4 + 1) * LDSP + ak] = (_Float16)av.y;
        ldsAT[(aq * 4 + 2) * LDSP + ak] = (_Float16)av.z;
        ldsAT[(aq * 4 + 3) * LDSP + ak] = (_Float16)av.w;
        __syncthreads();

        // ---- build WMMA fragments per the ISA 16-bit layouts (wave32)
        const int hsel = lane >> 4;   // lane group 0/1
        const int ml   = lane & 15;
        // A (16x32 f16): lanes 0-15 hold K [0..7]u[16..23]; lanes 16-31 hold +8
        const _Float16* zp0 = &ldsZ[(m_base + ml) * LDSP + hsel * 8];
        const _Float16* zp1 = &ldsZ[(m_base + 16 + ml) * LDSP + hsel * 8];
        const v8h a0lo = *reinterpret_cast<const v8h*>(zp0);
        const v8h a0hi = *reinterpret_cast<const v8h*>(zp0 + 16);
        const v8h a1lo = *reinterpret_cast<const v8h*>(zp1);
        const v8h a1hi = *reinterpret_cast<const v8h*>(zp1 + 16);
        // B (32x16 f16): lanes 0-15 hold K [0..7]u[16..23]; lanes 16-31 hold +8
        const _Float16* bp = &ldsAT[(n_base + ml) * LDSP + hsel * 8];
        const v8h blo = *reinterpret_cast<const v8h*>(bp);
        const v8h bhi = *reinterpret_cast<const v8h*>(bp + 16);
        v16h af0, af1, bf;
        #pragma unroll
        for (int i = 0; i < 8; ++i) {
            af0[i] = a0lo[i]; af0[i + 8] = a0hi[i];
            af1[i] = a1lo[i]; af1[i + 8] = a1hi[i];
            bf[i]  = blo[i];  bf[i + 8]  = bhi[i];
        }

        c0 = __builtin_amdgcn_wmma_f32_16x16x32_f16(
                 false, af0, false, bf, (short)0, c0, false, false);
        c1 = __builtin_amdgcn_wmma_f32_16x16x32_f16(
                 false, af1, false, bf, (short)0, c1, false, false);
        __syncthreads();
    }

    // ---- apply column normalization once to the 16 proj values of this lane
    const int n_local = n_base + (lane & 15);
    const float invc = invn[p_block + n_local];
    float pc[16];
    #pragma unroll
    for (int r = 0; r < 8; ++r) {
        pc[r]     = c0[r] * invc;
        pc[r + 8] = c1[r] * invc;
    }

    // ---- characteristic-function accumulation from registers.
    // C layout: VGPR r -> M = r (+8 for lanes 16-31), N = lane&15. Sums are
    // over M, so lane pairs (l, l+16) share a column: shfl_xor(16) merges them.
    for (int k = 0; k < KNOTS; ++k) {
        const float tk = (float)k * (3.0f / 16.0f);
        float cs = 0.0f, ss = 0.0f;
        #pragma unroll
        for (int r = 0; r < 16; ++r) {
            float s, co;
            __sincosf(pc[r] * tk, &s, &co);
            cs += co; ss += s;
        }
        cs += __shfl_xor(cs, 16, 32);
        ss += __shfl_xor(ss, 16, 32);
        if (lane < 16) {
            atomicAdd(&lds_cos[n_local * KNOTS + k], cs);
            atomicAdd(&lds_sin[n_local * KNOTS + k], ss);
        }
    }
    __syncthreads();

    // ---- flush block partials to L2-resident global accumulators
    for (int i = tid; i < TILE_N * KNOTS; i += 256) {
        const int nl = i / KNOTS, k = i % KNOTS;
        atomicAdd(&cosA[(size_t)(p_block + nl) * KNOTS + k], lds_cos[i]);
        atomicAdd(&sinA[(size_t)(p_block + nl) * KNOTS + k], lds_sin[i]);
    }
}

// ---------------------------------------------------------------- finalize
__global__ void sig_finalize_kernel(const float* __restrict__ cosA,
                                    const float* __restrict__ sinA,
                                    float* __restrict__ out) {
    const float dt = 3.0f / 16.0f;
    const float inv_n = 1.0f / (float)N_ROWS;
    float acc = 0.0f;
    for (int p = threadIdx.x; p < P_DIM; p += 256) {
        for (int k = 0; k < KNOTS; ++k) {
            const float t  = dt * (float)k;
            const float ph = expf(-0.5f * t * t);
            const float w  = ((k == 0) || (k == KNOTS - 1)) ? dt : 2.0f * dt;
            const float cm = cosA[(size_t)p * KNOTS + k] * inv_n;
            const float sm = sinA[(size_t)p * KNOTS + k] * inv_n;
            acc += ((cm - ph) * (cm - ph) + sm * sm) * (w * ph);
        }
    }
    __shared__ float red[256];
    red[threadIdx.x] = acc;
    __syncthreads();
    for (int s = 128; s > 0; s >>= 1) {
        if (threadIdx.x < s) red[threadIdx.x] += red[threadIdx.x + s];
        __syncthreads();
    }
    if (threadIdx.x == 0)
        out[0] = red[0] * (float)N_ROWS / (float)P_DIM;
}

// ---------------------------------------------------------------- launcher
extern "C" void kernel_launch(void* const* d_in, const int* in_sizes, int n_in,
                              void* d_out, int out_size, void* d_ws, size_t ws_size,
                              hipStream_t stream) {
    (void)in_sizes; (void)n_in; (void)out_size; (void)ws_size;
    const float* z = (const float*)d_in[0];
    const float* a = (const float*)d_in[1];

    float* invn = (float*)d_ws;                 // P_DIM
    float* cosA = invn + P_DIM;                 // P_DIM*KNOTS
    float* sinA = cosA + (size_t)P_DIM * KNOTS; // P_DIM*KNOTS
    float* out  = (float*)d_out;

    const int nz = P_DIM + 2 * P_DIM * KNOTS;
    sig_zero_kernel<<<(nz + 255) / 256, 256, 0, stream>>>(invn, nz);
    sig_colnorm_kernel<<<(P_DIM + 255) / 256, 256, 0, stream>>>(a, invn);

    dim3 grid(P_DIM / TILE_N, N_ROWS / TILE_M);  // (32, 64)
    sig_fused_kernel<<<grid, 256, 0, stream>>>(z, a, invn, cosA, sinA);

    sig_finalize_kernel<<<1, 256, 0, stream>>>(cosA, sinA, out);
}
